// AttentionModel_19679540150395
// MI455X (gfx1250) — matrix operand order repaired and checked by reading, not compiled, not run on hardware
//
#include <hip/hip_runtime.h>
#include <math.h>

typedef __attribute__((ext_vector_type(16))) __bf16 v16bf;
typedef __attribute__((ext_vector_type(8)))  __bf16 v8bf;
typedef __attribute__((ext_vector_type(8)))  float  v8f;
typedef __attribute__((ext_vector_type(4)))  float  v4f;

#define FEAT 512
#define HID  256

// ---------- fragment loaders (per CDNA5 ISA 7.12.2 bf16 layouts) ----------

// B-matrix 32x16 (KxN) bf16: lanes 0-15 hold K=kb*32+{0..7,16..23} of col N=lane,
// lanes 16-31 hold K=kb*32+{8..15,24..31} of col N=lane-16. Wt stored as [N][Kdim].
__device__ __forceinline__ v16bf load_bfrag(const __bf16* __restrict__ Wt,
                                            int Kdim, int n0, int kb, int lane) {
  const int n = n0 + (lane & 15);
  const int k = kb * 32 + ((lane & 16) ? 8 : 0);
  const __bf16* p = Wt + (size_t)n * Kdim + k;
  const v8bf lo = *(const v8bf*)(p);
  const v8bf hi = *(const v8bf*)(p + 16);
  v16bf r;
#pragma unroll
  for (int i = 0; i < 8; ++i) { r[i] = lo[i]; r[8 + i] = hi[i]; }
  return r;
}

// A-matrix 16x32 bf16 from an f32 row: lane = M; lanes 0-15 hold K ko=0:{0..7,16..23},
// lanes 16-31 hold ko=8:{8..15,24..31}. Elements 0..7 = K ko+0..7, 8..15 = K ko+16..23.
__device__ __forceinline__ v16bf load_afrag_x(const float* __restrict__ xrow,
                                              int kb, int lane) {
  const int ko = (lane & 16) ? 8 : 0;
  const float* p = xrow + kb * 32 + ko;
  const v4f f0 = *(const v4f*)(p);
  const v4f f1 = *(const v4f*)(p + 4);
  const v4f f2 = *(const v4f*)(p + 16);
  const v4f f3 = *(const v4f*)(p + 20);
  v16bf r;
#pragma unroll
  for (int i = 0; i < 4; ++i) {
    r[i]      = (__bf16)f0[i];
    r[4 + i]  = (__bf16)f1[i];
    r[8 + i]  = (__bf16)f2[i];
    r[12 + i] = (__bf16)f3[i];
  }
  return r;
}

// Same A layout but from a bf16 tile staged in LDS, row stride ldr elements.
__device__ __forceinline__ v16bf load_afrag_lds(const __bf16* hbase, int ldr,
                                                int kb, int lane) {
  const int m  = lane & 15;
  const int ko = (lane & 16) ? 8 : 0;
  const __bf16* p = hbase + m * ldr + kb * 32 + ko;
  const v8bf lo = *(const v8bf*)(p);        // K ko+0..7
  const v8bf hi = *(const v8bf*)(p + 16);   // K ko+16..23
  v16bf r;
#pragma unroll
  for (int i = 0; i < 8; ++i) { r[i] = lo[i]; r[8 + i] = hi[i]; }
  return r;
}

// ---------- kernel 1: weight transpose/convert + gf zero ----------
__global__ void __launch_bounds__(256)
prep_kernel(const float* __restrict__ W1, const float* __restrict__ Wa,
            const float* __restrict__ Wb,
            __bf16* __restrict__ W1t, __bf16* __restrict__ Wat,
            __bf16* __restrict__ Wbt, float* __restrict__ gf) {
  const int i = blockIdx.x * blockDim.x + threadIdx.x;
  if (i < FEAT * HID) {                 // W1 [512,256] -> W1t [256][512]
    const int k = i / HID, n = i % HID;
    W1t[(size_t)n * FEAT + k] = (__bf16)W1[i];
  }
  if (i < HID * HID) {                  // Wa/Wb [256,256] -> [256][256] transposed
    const int k = i / HID, n = i % HID;
    Wat[(size_t)n * HID + k] = (__bf16)Wa[i];
    Wbt[(size_t)n * HID + k] = (__bf16)Wb[i];
  }
  if (i < HID) gf[i] = 0.0f;
}

// ---------- kernel 2: h = relu(xW1+b1); s = (tanh(hWa+ba)*sigmoid(hWb+bb))@Wc + bc ----------
__global__ void __launch_bounds__(256)
pass1_kernel(const float* __restrict__ x,
             const float* __restrict__ b1, const float* __restrict__ ba,
             const float* __restrict__ bb, const float* __restrict__ Wc,
             const float* __restrict__ bc,
             const __bf16* __restrict__ W1t, const __bf16* __restrict__ Wat,
             const __bf16* __restrict__ Wbt,
             float* __restrict__ s_out, int N, int ntiles) {
  __shared__ __bf16 lds_h[8][16][HID];          // 64 KB: per-wave 16x256 bf16 tile
  const int lane = threadIdx.x & 31;
  const int wave = threadIdx.x >> 5;
  const int m    = lane & 15;
  const int half = lane >> 4;
  const int nwaves = gridDim.x * 8;
  __bf16 (*myh)[HID] = lds_h[wave];
  const float bcv = bc[0];

  for (int tile = blockIdx.x * 8 + wave; tile < ntiles; tile += nwaves) {
    const int row0 = tile * 16;
    int rowm = row0 + m; if (rowm >= N) rowm = N - 1;
    const float* xrow = x + (size_t)rowm * FEAT;

    // x A-fragments for the whole K=512, loaded once, reused for 16 col blocks
    v16bf xa[16];
#pragma unroll
    for (int kb = 0; kb < 16; ++kb) xa[kb] = load_afrag_x(xrow, kb, lane);

#pragma unroll 1
    for (int nb = 0; nb < 16; ++nb) {
      v8f acc = {};
#pragma unroll
      for (int kb = 0; kb < 16; ++kb) {
        const v16bf bf = load_bfrag(W1t, FEAT, nb * 16, kb, lane);
        acc = __builtin_amdgcn_wmma_f32_16x16x32_bf16(false, xa[kb], false, bf,
                                                      (short)0, acc, false, false);
      }
      const float bias = b1[nb * 16 + m];
#pragma unroll
      for (int v = 0; v < 8; ++v) {          // C layout: lane half selects M base
        float hv = acc[v] + bias;
        hv = hv > 0.0f ? hv : 0.0f;
        myh[v + 8 * half][nb * 16 + m] = (__bf16)hv;
      }
    }
    // Same-wave LDS RAW: DS ops are issued in order within a wave (no barrier needed).

    v16bf ha[8];
#pragma unroll
    for (int kb = 0; kb < 8; ++kb) ha[kb] = load_afrag_lds(&myh[0][0], HID, kb, lane);

    float sp[8];
#pragma unroll
    for (int v = 0; v < 8; ++v) sp[v] = 0.0f;

#pragma unroll 1
    for (int nb = 0; nb < 16; ++nb) {
      v8f acca = {}, accb = {};
#pragma unroll
      for (int kb = 0; kb < 8; ++kb) {
        const v16bf bfa = load_bfrag(Wat, HID, nb * 16, kb, lane);
        acca = __builtin_amdgcn_wmma_f32_16x16x32_bf16(false, ha[kb], false, bfa,
                                                       (short)0, acca, false, false);
        const v16bf bfb = load_bfrag(Wbt, HID, nb * 16, kb, lane);
        accb = __builtin_amdgcn_wmma_f32_16x16x32_bf16(false, ha[kb], false, bfb,
                                                       (short)0, accb, false, false);
      }
      const int col = nb * 16 + m;
      const float biasa = ba[col], biasb = bb[col], wc = Wc[col];
#pragma unroll
      for (int v = 0; v < 8; ++v) {
        const float av = tanhf(acca[v] + biasa);
        const float bv = 1.0f / (1.0f + expf(-(accb[v] + biasb)));
        sp[v] += av * bv * wc;
      }
    }

    // reduce over N (the 16 lanes within each half-wave)
#pragma unroll
    for (int v = 0; v < 8; ++v) {
      float t = sp[v];
      t += __shfl_xor(t, 1, 32);
      t += __shfl_xor(t, 2, 32);
      t += __shfl_xor(t, 4, 32);
      t += __shfl_xor(t, 8, 32);
      sp[v] = t;
    }
    if (m == 0) {
#pragma unroll
      for (int v = 0; v < 8; ++v) {
        const int r = row0 + v + 8 * half;
        if (r < N) s_out[r] = sp[v] + bcv;
      }
    }
  }
}

// ---------- kernel 3: softmax statistics (max, 1/sum-exp) ----------
__global__ void __launch_bounds__(1024)
stats_kernel(const float* __restrict__ s, float* __restrict__ stats, int N) {
  __shared__ float red[1024];
  const int t = threadIdx.x;
  float mx = -3.402823466e38f;
  for (int i = t; i < N; i += 1024) mx = fmaxf(mx, s[i]);
  red[t] = mx; __syncthreads();
  for (int off = 512; off > 0; off >>= 1) {
    if (t < off) red[t] = fmaxf(red[t], red[t + off]);
    __syncthreads();
  }
  mx = red[0]; __syncthreads();
  float sum = 0.0f;
  for (int i = t; i < N; i += 1024) sum += expf(s[i] - mx);
  red[t] = sum; __syncthreads();
  for (int off = 512; off > 0; off >>= 1) {
    if (t < off) red[t] += red[t + off];
    __syncthreads();
  }
  if (t == 0) { stats[0] = mx; stats[1] = 1.0f / red[0]; }
}

// ---------- kernel 4: A = softmax(s); global_feat += A*h (h recomputed via WMMA) ----------
__global__ void __launch_bounds__(256)
pass2_kernel(const float* __restrict__ x, const float* __restrict__ b1,
             const __bf16* __restrict__ W1t,
             const float* __restrict__ s, const float* __restrict__ stats,
             float* __restrict__ A_out, float* __restrict__ gf,
             int N, int ntiles) {
  const int lane = threadIdx.x & 31;
  const int wave = threadIdx.x >> 5;
  const int m    = lane & 15;
  const int half = lane >> 4;
  const int nwaves = gridDim.x * 8;
  const float mx = stats[0], inv = stats[1];

  float gfp[16];
#pragma unroll
  for (int i = 0; i < 16; ++i) gfp[i] = 0.0f;

  for (int tile = blockIdx.x * 8 + wave; tile < ntiles; tile += nwaves) {
    const int row0 = tile * 16;
    int rowm = row0 + m; if (rowm >= N) rowm = N - 1;
    const float* xrow = x + (size_t)rowm * FEAT;

    float aw[8];
#pragma unroll
    for (int v = 0; v < 8; ++v) {
      const int r = row0 + v + 8 * half;
      aw[v] = (r < N) ? expf(s[r] - mx) * inv : 0.0f;
    }
    if (half == 0) {
      const int r = row0 + m;
      if (r < N) A_out[r] = expf(s[r] - mx) * inv;
    }

    v16bf xa[16];
#pragma unroll
    for (int kb = 0; kb < 16; ++kb) xa[kb] = load_afrag_x(xrow, kb, lane);

#pragma unroll 1
    for (int nb = 0; nb < 16; ++nb) {
      v8f acc = {};
#pragma unroll
      for (int kb = 0; kb < 16; ++kb) {
        const v16bf bf = load_bfrag(W1t, FEAT, nb * 16, kb, lane);
        acc = __builtin_amdgcn_wmma_f32_16x16x32_bf16(false, xa[kb], false, bf,
                                                      (short)0, acc, false, false);
      }
      const float bias = b1[nb * 16 + m];
      float part = 0.0f;
#pragma unroll
      for (int v = 0; v < 8; ++v) {
        float hv = acc[v] + bias;
        hv = hv > 0.0f ? hv : 0.0f;
        part += hv * aw[v];
      }
      gfp[nb] += part;
    }
  }

#pragma unroll
  for (int nb = 0; nb < 16; ++nb) {
    float v = gfp[nb];
    v += __shfl_down(v, 16, 32);          // fold M=8..15 half onto M=0..7 half
    if (half == 0) atomicAdd(&gf[nb * 16 + m], v);
  }
}

// ---------- kernel 5: heads ----------
__global__ void __launch_bounds__(256)
head_kernel(const float* __restrict__ gf,
            const float* __restrict__ Wr, const float* __restrict__ br,
            const float* __restrict__ Wcls, const float* __restrict__ bcls,
            const float* __restrict__ Wreg, const float* __restrict__ breg,
            float* __restrict__ out, int N) {
  __shared__ float gfl[HID], hrl[HID];
  const int c = threadIdx.x;
  gfl[c] = gf[c];
  __syncthreads();
  out[N + c] = gfl[c];                                   // global_feat
  float acc = br[c];
  for (int k = 0; k < HID; ++k) acc += gfl[k] * Wr[k * HID + c];
  hrl[c] = acc > 0.0f ? acc : 0.0f;
  __syncthreads();
  if (c < 33) {
    float a = bcls[c];
    for (int k = 0; k < HID; ++k) a += hrl[k] * Wcls[k * 33 + c];
    out[N + HID + c] = a;                                // cls_logits
  }
  if (c < 55) {
    float a = breg[c];
    for (int k = 0; k < HID; ++k) a += hrl[k] * Wreg[k * 55 + c];
    out[N + HID + 33 + c] = a;                           // reg_values
  }
}

extern "C" void kernel_launch(void* const* d_in, const int* in_sizes, int n_in,
                              void* d_out, int out_size, void* d_ws, size_t ws_size,
                              hipStream_t stream) {
  (void)n_in; (void)out_size; (void)ws_size;
  const float* x    = (const float*)d_in[0];
  const float* W1   = (const float*)d_in[1];
  const float* b1   = (const float*)d_in[2];
  const float* Wa   = (const float*)d_in[3];
  const float* ba   = (const float*)d_in[4];
  const float* Wb   = (const float*)d_in[5];
  const float* bb   = (const float*)d_in[6];
  const float* Wc   = (const float*)d_in[7];
  const float* bc   = (const float*)d_in[8];
  const float* Wr   = (const float*)d_in[9];
  const float* br   = (const float*)d_in[10];
  const float* Wcls = (const float*)d_in[11];
  const float* bcls = (const float*)d_in[12];
  const float* Wreg = (const float*)d_in[13];
  const float* breg = (const float*)d_in[14];
  float* out = (float*)d_out;

  const int N = in_sizes[0] / FEAT;
  const int ntiles = (N + 15) / 16;

  char* ws = (char*)d_ws;
  __bf16* W1t = (__bf16*)(ws);
  __bf16* Wat = (__bf16*)(ws + (size_t)FEAT * HID * 2);
  __bf16* Wbt = (__bf16*)(ws + (size_t)FEAT * HID * 2 + (size_t)HID * HID * 2);
  size_t off = (size_t)FEAT * HID * 2 + 2 * (size_t)HID * HID * 2;   // 512 KiB
  float* s_buf = (float*)(ws + off);
  off += (((size_t)N * 4) + 255) & ~(size_t)255;
  float* stats = (float*)(ws + off);
  off += 256;
  float* gf = (float*)(ws + off);

  prep_kernel<<<(FEAT * HID + 255) / 256, 256, 0, stream>>>(W1, Wa, Wb, W1t, Wat, Wbt, gf);

  const int grid = (ntiles + 7) / 8;   // one 16-row tile per wave, 8 waves/WG
  pass1_kernel<<<grid, 256, 0, stream>>>(x, b1, ba, bb, Wc, bc, W1t, Wat, Wbt,
                                         s_buf, N, ntiles);
  stats_kernel<<<1, 1024, 0, stream>>>(s_buf, stats, N);
  pass2_kernel<<<grid, 256, 0, stream>>>(x, b1, W1t, s_buf, stats, out, gf, N, ntiles);
  head_kernel<<<1, 256, 0, stream>>>(gf, Wr, br, Wcls, bcls, Wreg, breg, out, N);
}
